// RNN_58832462021368
// MI455X (gfx1250) — matrix operand order repaired and checked by reading, not compiled, not run on hardware
//
#include <hip/hip_runtime.h>
#include <hip/hip_bf16.h>

#define SEQ_LEN 128
#define BATCH   64
#define VOCAB   10000
#define EMB     512
#define HID     512

typedef __bf16 bf16_t;
typedef __bf16 v16bf __attribute__((ext_vector_type(16)));
typedef __bf16 v8bf  __attribute__((ext_vector_type(8)));
typedef float  v8f   __attribute__((ext_vector_type(8)));

// round-to-nearest-even f32 -> bf16
__device__ __forceinline__ bf16_t f2bf(float f) {
  unsigned u = __builtin_bit_cast(unsigned, f);
  u += 0x7fffu + ((u >> 16) & 1u);
  unsigned short h = (unsigned short)(u >> 16);
  return __builtin_bit_cast(bf16_t, h);
}

__device__ __forceinline__ v8f wmma_bf16(v16bf a, v16bf b, v8f c) {
  // D = A(16x32 bf16) x B(32x16 bf16) + C(16x16 f32)
  return __builtin_amdgcn_wmma_f32_16x16x32_bf16(false, a, false, b, (short)0, c,
                                                 false, false);
}

// A fragment (16x32 bf16) for one lane from a row-major bf16 buffer
// (works for both LDS and global pointers -> ds_load_b128 / global_load_b128).
// rowbase = &buf[m * stride]; lane group g = lane>>4.
// elements 0..15 = K[kb + 16g .. +15]
__device__ __forceinline__ v16bf load_afrag(const bf16_t* rowbase, int kb, int g) {
  union { v16bf v; v8bf h[2]; } u;
  u.h[0] = *(const v8bf*)(rowbase + kb + 16 * g);
  u.h[1] = *(const v8bf*)(rowbase + kb + 16 * g + 8);
  return u.v;
}

// ---------------------------------------------------------------------------
// Weight prep: f32 [K][N] -> bf16 [N][K] (B fragments contiguous in K)
// ---------------------------------------------------------------------------
__global__ void transpose_f32_to_bf16(const float* __restrict__ in,
                                      bf16_t* __restrict__ out, int K, int N) {
  int total = K * N;
  for (int idx = blockIdx.x * blockDim.x + threadIdx.x; idx < total;
       idx += gridDim.x * blockDim.x) {
    int k = idx / N;
    int n = idx - k * N;                 // coalesced read along n
    out[(size_t)n * K + k] = f2bf(in[idx]);
  }
}

// ---------------------------------------------------------------------------
// Embedding gather (fully parallel, off the serial path):
// Xbf[t*64 + b][0..511] = bf16(emb[inputs[t][b]][:])
// ---------------------------------------------------------------------------
__global__ void gather_emb_bf16(const int* __restrict__ inputs,
                                const float* __restrict__ emb,
                                bf16_t* __restrict__ Xbf) {
  const int total4 = SEQ_LEN * BATCH * (EMB / 4);    // 1,048,576 float4
  for (int idx = blockIdx.x * blockDim.x + threadIdx.x; idx < total4;
       idx += gridDim.x * blockDim.x) {
    int row = idx >> 7;                              // 128 float4 per row
    int c4  = idx & 127;
    int tok = inputs[row];
    float4 v = *(const float4*)(emb + (size_t)tok * EMB + (size_t)c4 * 4);
    bf16_t* d = Xbf + (size_t)row * EMB + c4 * 4;
    d[0] = f2bf(v.x); d[1] = f2bf(v.y); d[2] = f2bf(v.z); d[3] = f2bf(v.w);
  }
}

// ---------------------------------------------------------------------------
// Serial recurrence: one workgroup (32 waves), hidden state in LDS.
// GEMM1 [64,1024]x[1024,512] -> h0n  (A: Xbf global | h0 LDS)
// GEMM2 [64,512]x[512,512]   -> x1   (A: h0n LDS)
// GEMM3 [64,1024]x[1024,512] -> h1n  (A: x1 LDS | h1 LDS), h1n -> H1 global
// Wave w owns N-tile w (cols w*16..w*16+15) and all 4 M-tiles.
// ---------------------------------------------------------------------------
__global__ void __launch_bounds__(1024, 1)
rnn_recurrence_kernel(const float* __restrict__ init_hidden,
                      const bf16_t* __restrict__ Xbf,    // [8192][512] bf16
                      const bf16_t* __restrict__ Wh0T,   // [512][1024] bf16
                      const float* __restrict__ bh0,
                      const bf16_t* __restrict__ Wh1T,   // [512][1024] bf16
                      const float* __restrict__ bh1,
                      const bf16_t* __restrict__ WfcT,   // [512][512]  bf16
                      const float* __restrict__ bfc,
                      bf16_t* __restrict__ H1,           // [8192][512] bf16
                      float* __restrict__ final_hidden)  // [2][64][512] f32
{
  extern __shared__ __align__(16) char smem[];
  bf16_t* x1buf = (bf16_t*)smem;          // [64][512]: x1
  bf16_t* h0buf = x1buf + BATCH * HID;    // [64][512]: h0
  bf16_t* h1buf = h0buf + BATCH * HID;    // [64][512]: h1

  const int tid  = threadIdx.x;
  const int lane = tid & 31;
  const int w    = tid >> 5;              // wave id == N-tile, 0..31
  const int g    = lane >> 4;
  const int nl   = lane & 15;
  const int n    = w * 16 + nl;           // output column, 0..511

  for (int j = tid; j < BATCH * HID; j += 1024) {
    h0buf[j] = f2bf(init_hidden[j]);
    h1buf[j] = f2bf(init_hidden[BATCH * HID + j]);
  }
  __syncthreads();

  const float bias0 = bh0[n];
  const float biasf = bfc[n];
  const float bias1 = bh1[n];

  for (int t = 0; t < SEQ_LEN; ++t) {
    // ---- GEMM1: h0n = tanh(concat(x_t, h0) @ Wh0 + bh0) ----
    {
      v8f acc[4] = {};
      const bf16_t* Xrow = Xbf + (size_t)t * BATCH * HID;
      for (int kb = 0; kb < HID; kb += 32) {             // K half 1: x_t (global)
        v16bf bf = *(const v16bf*)(Wh0T + (size_t)n * (2 * HID) + kb + 16 * g);
        #pragma unroll
        for (int mt = 0; mt < 4; ++mt) {
          v16bf af = load_afrag(Xrow + (size_t)(mt * 16 + nl) * HID, kb, g);
          acc[mt] = wmma_bf16(af, bf, acc[mt]);
        }
      }
      for (int kb = 0; kb < HID; kb += 32) {             // K half 2: h0 (LDS)
        v16bf bf =
            *(const v16bf*)(Wh0T + (size_t)n * (2 * HID) + HID + kb + 16 * g);
        #pragma unroll
        for (int mt = 0; mt < 4; ++mt) {
          v16bf af = load_afrag(h0buf + (mt * 16 + nl) * HID, kb, g);
          acc[mt] = wmma_bf16(af, bf, acc[mt]);
        }
      }
      __syncthreads();                    // everyone done reading h0buf
      #pragma unroll
      for (int mt = 0; mt < 4; ++mt) {
        #pragma unroll
        for (int r = 0; r < 8; ++r) {
          int m = mt * 16 + r + 8 * g;
          float v = tanhf(acc[mt][r] + bias0);
          h0buf[m * HID + n] = f2bf(v);
          if (t == SEQ_LEN - 1) final_hidden[m * HID + n] = v;   // layer 0
        }
      }
      __syncthreads();                    // h0buf ready
    }

    // ---- GEMM2: x1 = tanh(h0n @ Wfc0 + bfc0) ----
    {
      v8f acc[4] = {};
      for (int kb = 0; kb < HID; kb += 32) {
        v16bf bf = *(const v16bf*)(WfcT + (size_t)n * HID + kb + 16 * g);
        #pragma unroll
        for (int mt = 0; mt < 4; ++mt) {
          v16bf af = load_afrag(h0buf + (mt * 16 + nl) * HID, kb, g);
          acc[mt] = wmma_bf16(af, bf, acc[mt]);
        }
      }
      // x1buf's previous readers (last step's GEMM3) are past >=2 barriers
      #pragma unroll
      for (int mt = 0; mt < 4; ++mt) {
        #pragma unroll
        for (int r = 0; r < 8; ++r) {
          int m = mt * 16 + r + 8 * g;
          x1buf[m * HID + n] = f2bf(tanhf(acc[mt][r] + biasf));
        }
      }
      __syncthreads();                    // x1buf ready
    }

    // ---- GEMM3: h1n = tanh(concat(x1, h1) @ Wh1 + bh1) ----
    {
      v8f acc[4] = {};
      for (int kb = 0; kb < 2 * HID; kb += 32) {
        const bf16_t* abuf = (kb < HID) ? x1buf : h1buf;   // both LDS
        const int kk = kb & (HID - 1);
        v16bf bf = *(const v16bf*)(Wh1T + (size_t)n * (2 * HID) + kb + 16 * g);
        #pragma unroll
        for (int mt = 0; mt < 4; ++mt) {
          v16bf af = load_afrag(abuf + (mt * 16 + nl) * HID, kk, g);
          acc[mt] = wmma_bf16(af, bf, acc[mt]);
        }
      }
      __syncthreads();                    // everyone done reading h1buf/x1buf
      #pragma unroll
      for (int mt = 0; mt < 4; ++mt) {
        #pragma unroll
        for (int r = 0; r < 8; ++r) {
          int m = mt * 16 + r + 8 * g;
          float v = tanhf(acc[mt][r] + bias1);
          bf16_t hb = f2bf(v);
          h1buf[m * HID + n] = hb;
          H1[((size_t)t * BATCH + m) * HID + n] = hb;
          if (t == SEQ_LEN - 1)
            final_hidden[BATCH * HID + m * HID + n] = v;         // layer 1
        }
      }
      __syncthreads();                    // h1buf/x1buf safe to overwrite
    }
  }
}

// ---------------------------------------------------------------------------
// Logits: [8192,512](bf16) x [512,10000](bf16) + bo -> f32 d_out.
// 625 N-tiles (10000/16), 512 M-tiles; each wave: 4 M-tiles x 1 N-tile.
// 128 M-strips * 625 N-tiles = 80000 wave-jobs = 10000 blocks * 8 waves.
// Bound by the 327.7 MB output write; A/B stay L2-resident.
// ---------------------------------------------------------------------------
__global__ void __launch_bounds__(256)
logits_gemm_kernel(const bf16_t* __restrict__ H1,    // [8192][512]
                   const bf16_t* __restrict__ WoT,   // [10000][512]
                   const float* __restrict__ bo,
                   float* __restrict__ out)          // [8192][10000]
{
  const int lane = threadIdx.x & 31;
  const int wv   = threadIdx.x >> 5;
  const int g    = lane >> 4;
  const int nl   = lane & 15;

  const int job    = blockIdx.x * 8 + wv;            // 0..79999
  const int nt     = job % 625;
  const int mstrip = job / 625;                      // 0..127
  const int n      = nt * 16 + nl;
  const float bias = bo[n];

  v8f acc[4] = {};
  const bf16_t* bptr = WoT + (size_t)n * HID + 16 * g;
  for (int kb = 0; kb < HID; kb += 32) {
    v16bf bf = *(const v16bf*)(bptr + kb);
    #pragma unroll
    for (int i = 0; i < 4; ++i) {
      const bf16_t* arow = H1 + (size_t)((mstrip * 4 + i) * 16 + nl) * HID;
      acc[i] = wmma_bf16(load_afrag(arow, kb, g), bf, acc[i]);
    }
  }
  #pragma unroll
  for (int i = 0; i < 4; ++i) {
    #pragma unroll
    for (int r = 0; r < 8; ++r) {
      int m = (mstrip * 4 + i) * 16 + r + 8 * g;
      out[(size_t)m * VOCAB + n] = acc[i][r] + bias; // 16-lane contiguous cols
    }
  }
}

extern "C" void kernel_launch(void* const* d_in, const int* in_sizes, int n_in,
                              void* d_out, int out_size, void* d_ws, size_t ws_size,
                              hipStream_t stream) {
  (void)in_sizes; (void)n_in; (void)out_size; (void)ws_size;

  const int*   inputs = (const int*)  d_in[0];
  const float* init_h = (const float*)d_in[1];
  const float* emb    = (const float*)d_in[2];
  const float* Wh0    = (const float*)d_in[3];
  const float* bh0    = (const float*)d_in[4];
  const float* Wh1    = (const float*)d_in[5];
  const float* bh1    = (const float*)d_in[6];
  const float* Wfc0   = (const float*)d_in[7];
  const float* bfc0   = (const float*)d_in[8];
  const float* Wo     = (const float*)d_in[9];
  const float* bo     = (const float*)d_in[10];
  float* out = (float*)d_out;

  // workspace (bf16): WoT | Wh0T | Wh1T | WfcT | H1 | Xbf  (~28.3 MB)
  bf16_t* WoT  = (bf16_t*)d_ws;                       // [10000][512]
  bf16_t* Wh0T = WoT  + (size_t)VOCAB * HID;          // [512][1024]
  bf16_t* Wh1T = Wh0T + (size_t)2 * HID * HID;        // [512][1024]
  bf16_t* WfcT = Wh1T + (size_t)2 * HID * HID;        // [512][512]
  bf16_t* H1   = WfcT + (size_t)HID * HID;            // [8192][512]
  bf16_t* Xbf  = H1   + (size_t)SEQ_LEN * BATCH * HID;// [8192][512]

  transpose_f32_to_bf16<<<2048, 256, 0, stream>>>(Wo,   WoT,  HID,     VOCAB);
  transpose_f32_to_bf16<<<512,  256, 0, stream>>>(Wh0,  Wh0T, 2 * HID, HID);
  transpose_f32_to_bf16<<<512,  256, 0, stream>>>(Wh1,  Wh1T, 2 * HID, HID);
  transpose_f32_to_bf16<<<256,  256, 0, stream>>>(Wfc0, WfcT, HID,     HID);
  gather_emb_bf16<<<4096, 256, 0, stream>>>(inputs, emb, Xbf);

  const size_t logits_elems = (size_t)SEQ_LEN * BATCH * VOCAB;  // 81,920,000
  const size_t lds_bytes = (size_t)3 * BATCH * HID * sizeof(bf16_t); // 192 KB

  rnn_recurrence_kernel<<<1, 1024, lds_bytes, stream>>>(
      init_h, Xbf, Wh0T, bh0, Wh1T, bh1, WfcT, bfc0, H1, out + logits_elems);

  logits_gemm_kernel<<<10000, 256, 0, stream>>>(H1, WoT, bo, out);
}
